// PerforatedConv2d_24721831755815
// MI455X (gfx1250) — matrix operand order repaired and checked by reading, not compiled, not run on hardware
//
#include <hip/hip_runtime.h>

// ---------------------------------------------------------------------------
// PerforatedConv2d (perf=2,2) on MI455X / gfx1250:
//   y[n,co,:112,:112] = repeat2x2( conv3x3_stride2_pad1(x, w) + bias )
// Split-precision bf16 WMMA implicit GEMM (hi*hi + hi*lo + lo*hi, f32 acc):
// 3x v_wmma_f32_16x16x32_bf16 per K=32 step -> compute ~= 17.6us HBM floor.
// 3-tap weight tiles (69 KB LDS -> multi-block/WGP occupancy), async
// global->LDS weight staging (ASYNCcnt), global_prefetch of next ci chunk.
// ---------------------------------------------------------------------------

typedef __attribute__((ext_vector_type(16))) __bf16 v16bf;
typedef __attribute__((ext_vector_type(8)))  float  v8f;
typedef int v4i __attribute__((vector_size(4 * sizeof(int))));

#define NB   32    // batch
#define CH   128   // channels (in == out)
#define HW   112   // input H == W
#define OHW  56    // subsampled conv output H == W
#define KCH  32    // ci chunk (one WMMA K)
#define NCC  4     // 128 / 32 chunks
#define PIH  9     // patch rows  (2*4 + 1 halo rows for 4 oh, k=3, s=2)
#define PIW  17    // patch cols  (2*8 + 1 halo for 8 ow)
#define WCHUNK (9*128*32)     // bf16 elems per weight ci-chunk plane in ws
#define WROW   (3*128*32)     // bf16 elems per 3-tap (one kh row) LDS tile

#define AS1 __attribute__((address_space(1)))
#define AS3 __attribute__((address_space(3)))

#if defined(__has_builtin)
#if __has_builtin(__builtin_amdgcn_global_load_async_to_lds_b128)
#define HAVE_ASYNC_LDS 1
#endif
#if __has_builtin(__builtin_amdgcn_s_wait_asynccnt)
#define HAVE_WAIT_ASYNC 1
#endif
#endif

__device__ __forceinline__ void wait_asynccnt0() {
#if defined(HAVE_WAIT_ASYNC)
  __builtin_amdgcn_s_wait_asynccnt(0);
#elif defined(HAVE_ASYNC_LDS)
  asm volatile("s_wait_asynccnt 0" ::: "memory");
#endif
}

// --- pre-kernel: split fp32 weights into bf16 hi/lo planes, transposed to
//     [cc][tap][co][ci32] so the main kernel bulk-copies rows to LDS.
__global__ __launch_bounds__(256) void split_weights(
    const float* __restrict__ w, unsigned short* __restrict__ hi,
    unsigned short* __restrict__ lo) {
  int e = blockIdx.x * 256 + threadIdx.x;          // 128*128*9 = 147456
  int tap = e % 9;
  int ci  = (e / 9) % CH;
  int co  = e / (9 * CH);
  float f = w[e];                                  // w[co][ci][kh][kw]
  unsigned u  = __float_as_uint(f);
  unsigned uh = u & 0xFFFF0000u;
  float r = f - __uint_as_float(uh);
  int cc = ci >> 5, cl = ci & 31;
  int o  = ((cc * 9 + tap) * CH + co) * KCH + cl;
  hi[o] = (unsigned short)(uh >> 16);
  lo[o] = (unsigned short)(__float_as_uint(r) >> 16);
}

__global__ __launch_bounds__(256) void perf_conv2d(
    const float* __restrict__ x, const unsigned short* __restrict__ wsHi,
    const unsigned short* __restrict__ wsLo, const float* __restrict__ bias,
    float* __restrict__ out) {
  // LDS: one kh-row (3 taps) of weights + input patch, bf16 hi/lo planes.
  __shared__ __align__(16) unsigned short sWhi[WROW];            // 24576 B
  __shared__ __align__(16) unsigned short sWlo[WROW];            // 24576 B
  __shared__ __align__(16) unsigned short sPhi[PIH * PIW * KCH]; //  9792 B
  __shared__ __align__(16) unsigned short sPlo[PIH * PIW * KCH]; //  9792 B

  const int tid  = threadIdx.x;
  const int lane = tid & 31;
  const int wv   = tid >> 5;          // 8 waves: wave wv owns co rows 16wv..
  const int n16  = lane & 15;
  const int khi  = lane >> 4;         // half-wave selector for WMMA layouts

  const int n   = blockIdx.z;
  const int oh0 = blockIdx.y * 4;     // 4 x 8 tile of 56x56 conv outputs
  const int ow0 = blockIdx.x * 8;

  // A k order used here: lanes 0-15 hold K 0..15, lanes 16-31 K 16..31
  // (matching B); dword j packs K pair (kb, kb+1).
  int kbA[8], kbB[8];
#pragma unroll
  for (int j = 0; j < 8; ++j) {
    kbA[j] = 2 * j + 16 * khi;
    kbB[j] = 2 * j + 16 * khi;        // B: lanes 0-15 K 0..15, 16-31 K 16..31
  }

  v8f acc[2] = {{}, {}};              // two 16-col N tiles -> 32 pixels

  for (int cc = 0; cc < NCC; ++cc) {
    __syncthreads();                  // previous chunk fully consumed

    // -- stage input patch: load fp32, split to bf16 hi/lo, ci innermost;
    //    prefetch next chunk's patch lines while we are at it.
    for (int e = tid; e < PIH * PIW * KCH; e += 256) {
      int iw = e % PIW;
      int ih = (e / PIW) % PIH;
      int ci = e / (PIH * PIW);
      int ihg = 2 * oh0 - 1 + ih;
      int iwg = 2 * ow0 - 1 + iw;
      float f = 0.0f;
      if (ihg >= 0 && ihg < HW && iwg >= 0 && iwg < HW) {
        size_t gi = (((size_t)n * CH + cc * KCH + ci) * HW + ihg) * HW + iwg;
        f = x[gi];
        if (cc + 1 < NCC)
          __builtin_prefetch(&x[gi + (size_t)KCH * HW * HW], 0, 3);
      }
      unsigned u  = __float_as_uint(f);
      unsigned uh = u & 0xFFFF0000u;
      float r = f - __uint_as_float(uh);
      int o = (ih * PIW + iw) * KCH + ci;
      sPhi[o] = (unsigned short)(uh >> 16);
      sPlo[o] = (unsigned short)(__float_as_uint(r) >> 16);
    }

    for (int kh = 0; kh < 3; ++kh) {
      if (kh) __syncthreads();        // previous kh-row tile consumed
      // -- stage one kh row (3 taps) of pre-split bf16 weights: pure 16B
      //    contiguous copies -> async global->LDS (tracked by ASYNCcnt).
      const uint4* srcH = (const uint4*)(wsHi + cc * WCHUNK + kh * WROW);
      const uint4* srcL = (const uint4*)(wsLo + cc * WCHUNK + kh * WROW);
      uint4* dH = (uint4*)sWhi;
      uint4* dL = (uint4*)sWlo;
#if defined(HAVE_ASYNC_LDS)
      for (int i = tid; i < WROW / 8; i += 256) {
        __builtin_amdgcn_global_load_async_to_lds_b128(
            (AS1 v4i*)(srcH + i), (AS3 v4i*)(dH + i), 0, 0);
        __builtin_amdgcn_global_load_async_to_lds_b128(
            (AS1 v4i*)(srcL + i), (AS3 v4i*)(dL + i), 0, 0);
      }
      wait_asynccnt0();
#else
      for (int i = tid; i < WROW / 8; i += 256) { dH[i] = srcH[i]; dL[i] = srcL[i]; }
#endif
      __syncthreads();

      for (int kw = 0; kw < 3; ++kw) {
        union { unsigned u[8]; v16bf v; } Ah, Al;
        const int abase = (kw * CH + wv * 16 + n16) * KCH;
#pragma unroll
        for (int j = 0; j < 8; ++j) {
          Ah.u[j] = *(const unsigned*)(sWhi + abase + kbA[j]);
          Al.u[j] = *(const unsigned*)(sWlo + abase + kbA[j]);
        }
#pragma unroll
        for (int t = 0; t < 2; ++t) {
          const int p   = t * 16 + n16;                    // pixel in 4x8 tile
          const int pos = (((p >> 3) * 2 + kh) * PIW + ((p & 7) * 2 + kw)) * KCH;
          union { unsigned u[8]; v16bf v; } Bh, Bl;
#pragma unroll
          for (int j = 0; j < 8; ++j) {
            Bh.u[j] = *(const unsigned*)(sPhi + pos + kbB[j]);
            Bl.u[j] = *(const unsigned*)(sPlo + pos + kbB[j]);
          }
          acc[t] = __builtin_amdgcn_wmma_f32_16x16x32_bf16(
              false, Ah.v, false, Bh.v, (short)0, acc[t], false, false);
          acc[t] = __builtin_amdgcn_wmma_f32_16x16x32_bf16(
              false, Ah.v, false, Bl.v, (short)0, acc[t], false, false);
          acc[t] = __builtin_amdgcn_wmma_f32_16x16x32_bf16(
              false, Al.v, false, Bh.v, (short)0, acc[t], false, false);
        }
      }
    }
  }

  // -- epilogue: bias + fused 2x2 nearest-neighbor duplication to 112x112.
  // C/D layout: VGPR v -> M = v + 8*khi, N = lane%16.
#pragma unroll
  for (int t = 0; t < 2; ++t) {
    const int p  = t * 16 + n16;
    const int oh = oh0 + (p >> 3);
    const int ow = ow0 + (p & 7);
#pragma unroll
    for (int v = 0; v < 8; ++v) {
      const int co = wv * 16 + khi * 8 + v;
      const float val = acc[t][v] + bias[co];
      const float2 dup = make_float2(val, val);
      const size_t base =
          (((size_t)n * CH + co) * HW + 2 * oh) * HW + 2 * ow;
      *(float2*)(out + base)      = dup;
      *(float2*)(out + base + HW) = dup;
    }
  }
}

extern "C" void kernel_launch(void* const* d_in, const int* in_sizes, int n_in,
                              void* d_out, int out_size, void* d_ws, size_t ws_size,
                              hipStream_t stream) {
  const float* x    = (const float*)d_in[0];   // 32x128x112x112 f32
  const float* w    = (const float*)d_in[1];   // 128x128x3x3    f32
  const float* bias = (const float*)d_in[2];   // 128            f32
  float* out = (float*)d_out;                  // 32x128x112x112 f32

  unsigned short* wsHi = (unsigned short*)d_ws;          // 294912 B
  unsigned short* wsLo = wsHi + NCC * WCHUNK;            // 294912 B (589824 total)

  split_weights<<<(CH * CH * 9) / 256, 256, 0, stream>>>(w, wsHi, wsLo);

  dim3 grid(OHW / 8, OHW / 4, NB);   // (7, 14, 32)
  perf_conv2d<<<grid, 256, 0, stream>>>(x, wsHi, wsLo, bias, out);
}
